// Attention_26637387170241
// MI455X (gfx1250) — matrix operand order repaired and checked by reading, not compiled, not run on hardware
//
#include <hip/hip_runtime.h>
#include <hip/hip_bf16.h>

// Complex attention, MI455X (gfx1250, wave32, WMMA). N=4, L=S=1024, H=8, D=64.
// f32 in/out.  Pre-pass converts K (elementwise) and V (transposed to (n,h,d,s))
// to f16 in d_ws; main kernel runs f16 WMMA with f32 accumulate.

#define CN 4
#define CL 1024
#define CH 8
#define CD 64
#define CS 1024
#define CHD (CH * CD)              // 512 floats between successive l (or s) rows
#define NLHD (CN * CL * CH * CD)   // 2,097,152
#define NLS  (CN * CL * CS)        // 4,194,304

typedef __attribute__((ext_vector_type(16))) _Float16 v16h;
typedef __attribute__((ext_vector_type(8)))  _Float16 v8h;
typedef __attribute__((ext_vector_type(8)))  float    v8f;

__device__ __forceinline__ v8f wmma_f16(v16h a, v16h b, v8f c) {
  // 8 args: (neg_a, A, neg_b, B, c_mod, C, reuse_a, reuse_b)
  return __builtin_amdgcn_wmma_f32_16x16x32_f16(false, a, false, b, (short)0, c,
                                                false, false);
}

// ---------- pre-pass 1: K f32 -> f16, same (n,l,h,d) layout -----------------
__global__ void __launch_bounds__(256)
convert_k_kernel(const float* __restrict__ kr, const float* __restrict__ ki,
                 _Float16* __restrict__ k16r, _Float16* __restrict__ k16i) {
  const int i = blockIdx.x * 256 + threadIdx.x;
  if (i < NLHD) {
    k16r[i] = (_Float16)kr[i];
    k16i[i] = (_Float16)ki[i];
  }
}

// ---------- pre-pass 2: V f32 (n,s,h,d) -> f16 transposed (n,h,d,s) ---------
// lane <-> d so the f32 reads are 128B-coalesced per wave; each thread writes
// one contiguous 32B run of 16 s-values.
__global__ void __launch_bounds__(256)
transpose_v_kernel(const float* __restrict__ vr, const float* __restrict__ vi,
                   _Float16* __restrict__ vtr, _Float16* __restrict__ vti) {
  const int g    = blockIdx.x * 256 + threadIdx.x;  // 131072 total
  const int d    = g & 63;
  const int sblk = (g >> 6) & 63;
  const int h    = (g >> 12) & 7;
  const int n    = g >> 15;
  const int s0   = sblk * 16;
  v16h r, im;
  for (int j = 0; j < 16; ++j) {
    const size_t src = ((size_t)(n * CL + s0 + j) * CH + h) * CD + d;
    r[j]  = (_Float16)vr[src];
    im[j] = (_Float16)vi[src];
  }
  const size_t dst = ((size_t)((n * CH + h) * CD) + d) * CS + s0;
  *(v16h*)(vtr + dst) = r;
  *(v16h*)(vti + dst) = im;
}

// ---------- main fused attention kernel -------------------------------------
__global__ void __launch_bounds__(256, 1)
cattn_kernel(const float* __restrict__ qr, const float* __restrict__ qi,
             const _Float16* __restrict__ k16r, const _Float16* __restrict__ k16i,
             const _Float16* __restrict__ vtr, const _Float16* __restrict__ vti,
             float* __restrict__ out)
{
  extern __shared__ float lds[];
  float*    sreal = lds;                          // 16*1024 f32 scores (real)
  float*    simag = sreal + 16 * 1024;            // 16*1024 f32 scores (imag)
  float*    amr   = simag + 16 * 1024;            // 16*1024 f32 head-mean accum
  float*    ami   = amr   + 16 * 1024;            // 16*1024 f32 head-mean accum
  _Float16* w16r  = (_Float16*)(ami + 16 * 1024); // 16*1024 f16 weights (real)
  _Float16* w16i  = w16r + 16 * 1024;             // 16*1024 f16 weights (imag)
  // total = 4*64KB + 2*32KB = 320 KB (full WGP budget)

  const int tid  = threadIdx.x;
  const int lane = tid & 31;
  const int wave = tid >> 5;          // 0..7
  const int lgrp = lane >> 4;         // 0/1 lane half
  const int lsub = lane & 15;

  const int n  = blockIdx.x >> 6;          // batch
  const int m0 = (blockIdx.x & 63) * 16;   // query-row tile base

  float* uout_r = out;
  float* uout_i = out + (size_t)NLHD;
  float* aout_r = out + (size_t)2 * NLHD;
  float* aout_i = aout_r + (size_t)NLS;

  // zero head-mean accumulators (amr, ami contiguous)
  for (int i = tid; i < 2 * 16 * 1024; i += 256) amr[i] = 0.0f;
  __syncthreads();

  for (int h = 0; h < CH; ++h) {
    // ============ Phase 1: scores S = scale * (Q K^T), complex =============
    // A-fragment (16x32 f16): lane m = lsub; halves[0..7] -> K=16*lgrp+0..7,
    // halves[8..15] -> K=16*lgrp+8+0..7.  Two k-steps cover D=64.
    v16h aQr[2], aQi[2], anQi[2];
    {
      const float* qbr = qr + ((size_t)(n * CL + m0 + lsub) * CH + h) * CD;
      const float* qbi = qi + ((size_t)(n * CL + m0 + lsub) * CH + h) * CD;
      for (int ks = 0; ks < 2; ++ks) {
        const int d0 = 32 * ks + 16 * lgrp;
        v16h fr, fi, fni;
        for (int t = 0; t < 8; ++t) {
          float r0 = qbr[d0 + t],      i0 = qbi[d0 + t];
          float r1 = qbr[d0 + 8 + t], i1 = qbi[d0 + 8 + t];
          fr[t]     = (_Float16)r0;  fi[t]     = (_Float16)i0;  fni[t]     = (_Float16)(-i0);
          fr[8 + t] = (_Float16)r1;  fi[8 + t] = (_Float16)i1;  fni[8 + t] = (_Float16)(-i1);
        }
        aQr[ks] = fr; aQi[ks] = fi; anQi[ks] = fni;
      }
    }

    // B-fragment (32x16 f16): lane col = key s0+lsub; halves[i] -> d =
    // 32*ks + 16*lgrp + i : one contiguous 32B f16 vector load per fragment.
    for (int st = wave; st < 64; st += 8) {      // waves split the 64 S-tiles
      const int s0 = st * 16;
      const _Float16* kbr = k16r + ((size_t)(n * CL + s0 + lsub) * CH + h) * CD + 16 * lgrp;
      const _Float16* kbi = k16i + ((size_t)(n * CL + s0 + lsub) * CH + h) * CD + 16 * lgrp;
      v8f cr = {}; v8f ci = {};
      for (int ks = 0; ks < 2; ++ks) {
        const v16h bKr = *(const v16h*)(kbr + 32 * ks);
        const v16h bKi = *(const v16h*)(kbi + 32 * ks);
        cr = wmma_f16(aQr[ks],  bKr, cr);   // + Qr.Kr
        cr = wmma_f16(anQi[ks], bKi, cr);   // - Qi.Ki
        ci = wmma_f16(aQr[ks],  bKi, ci);   // + Qr.Ki
        ci = wmma_f16(aQi[ks],  bKr, ci);   // + Qi.Kr
      }
      // C layout: VGPR r -> M = r + 8*lgrp, column = lsub
      for (int r = 0; r < 8; ++r) {
        const int M = 8 * lgrp + r;
        sreal[M * 1024 + s0 + lsub] = 0.125f * cr[r];   // scale = 1/sqrt(64)
        simag[M * 1024 + s0 + lsub] = 0.125f * ci[r];
      }
    }
    __syncthreads();

    // ============ Phase 2: row softmax -> f16 weights + head-mean ==========
    {
      const int row = tid >> 4;       // 16 threads per row
      const int p0  = (tid & 15) * 2; // column pairs, stride 32
      for (int comp = 0; comp < 2; ++comp) {
        float*    base = (comp ? simag : sreal) + row * 1024;
        float*    amb  = (comp ? ami   : amr)   + row * 1024;
        _Float16* wout = (comp ? w16i  : w16r)  + row * 1024;
        float mx = -3.402823466e38f;
        for (int c = p0; c < 1024; c += 32)
          mx = fmaxf(mx, fmaxf(base[c], base[c + 1]));
        for (int off = 8; off; off >>= 1) mx = fmaxf(mx, __shfl_xor(mx, off, 16));
        float sum = 0.0f;
        for (int c = p0; c < 1024; c += 32) {
          float e0 = __expf(base[c] - mx);
          float e1 = __expf(base[c + 1] - mx);
          base[c] = e0; base[c + 1] = e1;
          sum += e0 + e1;
        }
        for (int off = 8; off; off >>= 1) sum += __shfl_xor(sum, off, 16);
        const float inv = 1.0f / sum;
        for (int c = p0; c < 1024; c += 32) {
          float w0 = base[c] * inv;
          float w1 = base[c + 1] * inv;
          wout[c]     = (_Float16)w0;
          wout[c + 1] = (_Float16)w1;
          amb[c]     += w0 * 0.125f;         // mean over H=8 heads
          amb[c + 1] += w1 * 0.125f;
        }
      }
    }
    __syncthreads();

    // ============ Phase 3: U = W x V (complex), 16x1024 * 1024x64 ==========
    // even waves: real part  (cA = Wr.Vr, cB = Wi.Vi, U_r = cA - cB)
    // odd  waves: imag part  (cA = Wr.Vi, cB = Wi.Vr, U_i = cA + cB)
    // ntile = wave>>1 selects the 16-wide d block; full S per wave -> no
    // cross-wave reduction, no extra barrier.
    {
      const int ntile = wave >> 1;
      const int imagP = wave & 1;
      const int c0    = ntile * 16;
      const _Float16* vbA = (imagP ? vti : vtr) +
          ((size_t)((n * CH + h) * CD) + c0 + lsub) * CS + 16 * lgrp;
      const _Float16* vbB = (imagP ? vtr : vti) +
          ((size_t)((n * CH + h) * CD) + c0 + lsub) * CS + 16 * lgrp;
      v8f cA = {}; v8f cB = {};
      for (int ks = 0; ks < 32; ++ks) {          // 32 * 32 = full S = 1024
        const int s0 = ks * 32;
        // A fragments straight from f16 LDS (two 16B ds loads each)
        const _Float16* wrb = w16r + lsub * 1024 + s0 + 16 * lgrp;
        const _Float16* wib = w16i + lsub * 1024 + s0 + 16 * lgrp;
        const v8h r_lo = *(const v8h*)wrb;
        const v8h r_hi = *(const v8h*)(wrb + 8);
        const v8h i_lo = *(const v8h*)wib;
        const v8h i_hi = *(const v8h*)(wib + 8);
        const v16h awr = __builtin_shufflevector(r_lo, r_hi,
            0,1,2,3,4,5,6,7,8,9,10,11,12,13,14,15);
        const v16h awi = __builtin_shufflevector(i_lo, i_hi,
            0,1,2,3,4,5,6,7,8,9,10,11,12,13,14,15);
        // B fragments: one contiguous 32B f16 vector load each (V transposed)
        const v16h bA = *(const v16h*)(vbA + s0);
        const v16h bB = *(const v16h*)(vbB + s0);
        cA = wmma_f16(awr, bA, cA);
        cB = wmma_f16(awi, bB, cB);
      }
      float* uo = imagP ? uout_i : uout_r;
      const float sgn = imagP ? 1.0f : -1.0f;
      for (int r = 0; r < 8; ++r) {
        const int M = 8 * lgrp + r;
        const size_t off = ((size_t)(n * CL + m0 + M) * CH + h) * CD + c0 + lsub;
        uo[off] = cA[r] + sgn * cB[r];
      }
    }
    // no barrier needed here: next head's Phase-1 sync protects w16/scores
  } // heads

  // ============ write head-mean attention maps ============================
  __syncthreads();
  for (int i = tid; i < 16 * 1024; i += 256) {
    const int row = i >> 10;
    const int c   = i & 1023;
    const size_t off = (size_t)n * CL * CS + (size_t)(m0 + row) * CS + c;
    aout_r[off] = amr[i];
    aout_i[off] = ami[i];
  }
}

extern "C" void kernel_launch(void* const* d_in, const int* in_sizes, int n_in,
                              void* d_out, int out_size, void* d_ws, size_t ws_size,
                              hipStream_t stream) {
  (void)in_sizes; (void)n_in; (void)out_size; (void)ws_size;
  const float* qr = (const float*)d_in[0];
  const float* qi = (const float*)d_in[1];
  const float* kr = (const float*)d_in[2];
  const float* ki = (const float*)d_in[3];
  const float* vr = (const float*)d_in[4];
  const float* vi = (const float*)d_in[5];
  float* out = (float*)d_out;

  // workspace layout: 4 f16 arrays of NLHD halfs = 16 MB total
  _Float16* k16r = (_Float16*)d_ws;
  _Float16* k16i = k16r + (size_t)NLHD;
  _Float16* vtr  = k16i + (size_t)NLHD;
  _Float16* vti  = vtr  + (size_t)NLHD;

  // pre-pass: convert K, transpose+convert V (stream-ordered before main)
  convert_k_kernel<<<dim3(NLHD / 256), dim3(256), 0, stream>>>(kr, ki, k16r, k16i);
  transpose_v_kernel<<<dim3((CN * CH * CD * (CS / 16)) / 256), dim3(256), 0, stream>>>(
      vr, vi, vtr, vti);

  const size_t shmem = (size_t)(4 * 16 * 1024) * sizeof(float)
                     + (size_t)(2 * 16 * 1024) * sizeof(_Float16);   // 320 KB
  (void)hipFuncSetAttribute((const void*)cattn_kernel,
                            hipFuncAttributeMaxDynamicSharedMemorySize,
                            (int)shmem);
  cattn_kernel<<<dim3(CN * (CL / 16)), dim3(256), shmem, stream>>>(
      qr, qi, k16r, k16i, vtr, vti, out);
}
